// PCTPositionEmbedding_11982958756081
// MI455X (gfx1250) — matrix operand order repaired and checked by reading, not compiled, not run on hardware
//
#include <hip/hip_runtime.h>
#include <hip/hip_bf16.h>

typedef __attribute__((ext_vector_type(16))) _Float16 v16h;
typedef __attribute__((ext_vector_type(8)))  _Float16 v8h;
typedef __attribute__((ext_vector_type(8)))  float    v8f;

#define BB 8
#define CC 256
#define DD 64
#define NP 2048
#define BN_EPS 1e-5f

// Async DMA: global -> LDS, 16B per lane, tracked by ASYNCcnt.
// VDST operand = LDS byte offset (low 32 bits of the flat shared pointer),
// VADDR = 64-bit global address in a VGPR pair.
__device__ __forceinline__ void async_ld_b128(unsigned lds_off, const void* gptr) {
  unsigned long long ga = (unsigned long long)(uintptr_t)gptr;
  asm volatile("global_load_async_to_lds_b128 %0, %1, off"
               :: "v"(lds_off), "v"(ga) : "memory");
}
__device__ __forceinline__ void wait_asynccnt0() {
  asm volatile("s_wait_asynccnt 0x0" ::: "memory");
}

// ------------------------- elementwise kernels -------------------------

__global__ __launch_bounds__(256) void k_cvt_f16(const float* __restrict__ src,
                                                 _Float16* __restrict__ dst, int n) {
  int i = blockIdx.x * 256 + threadIdx.x;
  if (i < n) dst[i] = (_Float16)src[i];
}

// x [B,C,N] f32 -> xT [B,N,C] f16
__global__ __launch_bounds__(256) void k_transpose_cvt(const float* __restrict__ src,
                                                       _Float16* __restrict__ dst) {
  long long i = (long long)blockIdx.x * 256 + threadIdx.x;
  if (i >= (long long)BB * CC * NP) return;
  int n = (int)(i % NP);
  long long t = i / NP;
  int c = (int)(t % CC);
  int b = (int)(t / CC);
  dst[((long long)b * NP + n) * CC + c] = (_Float16)src[i];
}

// pos[b,c,n] = sum_j convpos_w[c,j] * xyz[b,n,j]
__global__ __launch_bounds__(256) void k_pos(const float* __restrict__ w,
                                             const float* __restrict__ xyz,
                                             float* __restrict__ pos) {
  long long i = (long long)blockIdx.x * 256 + threadIdx.x;
  if (i >= (long long)BB * CC * NP) return;
  int n = (int)(i % NP);
  long long t = i / NP;
  int c = (int)(t % CC);
  int b = (int)(t / CC);
  const float* xp = xyz + ((long long)b * NP + n) * 3;
  const float* wp = w + c * 3;
  pos[i] = wp[0] * xp[0] + wp[1] * xp[1] + wp[2] * xp[2];
}

// h2T[b,n,c] = f16(h[b,c,n] + pos[b,c,n])
__global__ __launch_bounds__(256) void k_h2T(const float* __restrict__ h,
                                             const float* __restrict__ pos,
                                             _Float16* __restrict__ h2T) {
  long long i = (long long)blockIdx.x * 256 + threadIdx.x;
  if (i >= (long long)BB * CC * NP) return;
  int n = (int)(i % NP);
  long long t = i / NP;
  int c = (int)(t % CC);
  int b = (int)(t / CC);
  h2T[((long long)b * NP + n) * CC + c] = (_Float16)(h[i] + pos[i]);
}

// per-row (n) softmax stats over energy[b][n][:]
__global__ __launch_bounds__(256) void k_rowstat(const float* __restrict__ e,
                                                 float* __restrict__ rmax,
                                                 float* __restrict__ rsum) {
  const float* row = e + ((long long)blockIdx.y * NP + blockIdx.x) * NP;
  __shared__ float red[256];
  int tid = threadIdx.x;
  float mx = -3.4e38f;
  for (int j = tid; j < NP; j += 256) mx = fmaxf(mx, row[j]);
  red[tid] = mx; __syncthreads();
  for (int s = 128; s > 0; s >>= 1) {
    if (tid < s) red[tid] = fmaxf(red[tid], red[tid + s]);
    __syncthreads();
  }
  mx = red[0]; __syncthreads();
  float sm = 0.0f;
  for (int j = tid; j < NP; j += 256) sm += __expf(row[j] - mx);
  red[tid] = sm; __syncthreads();
  for (int s = 128; s > 0; s >>= 1) {
    if (tid < s) red[tid] += red[tid + s];
    __syncthreads();
  }
  if (tid == 0) {
    long long idx = (long long)blockIdx.y * NP + blockIdx.x;
    rmax[idx] = mx;
    rsum[idx] = 1.0f / red[0];
  }
}

// per-column (m): colsum of row-softmaxed att, then write attT[b][m][n] (f16)
__global__ __launch_bounds__(256) void k_colnorm(const float* __restrict__ e,
                                                 const float* __restrict__ rmax,
                                                 const float* __restrict__ rsum,
                                                 _Float16* __restrict__ attT) {
  int m = blockIdx.x;
  int b = blockIdx.y;
  const float* eb = e + (long long)b * NP * NP;
  const float* mx = rmax + (long long)b * NP;
  const float* rs = rsum + (long long)b * NP;
  __shared__ float red[256];
  int tid = threadIdx.x;
  float sm = 0.0f;
  for (int n = tid; n < NP; n += 256)
    sm += __expf(eb[(long long)n * NP + m] - mx[n]) * rs[n];
  red[tid] = sm; __syncthreads();
  for (int s = 128; s > 0; s >>= 1) {
    if (tid < s) red[tid] += red[tid + s];
    __syncthreads();
  }
  float inv = 1.0f / (1e-9f + red[0]);
  _Float16* orow = attT + ((long long)b * NP + m) * NP;
  for (int n = tid; n < NP; n += 256)
    orow[n] = (_Float16)(__expf(eb[(long long)n * NP + m] - mx[n]) * rs[n] * inv);
}

// ------------------------- WMMA GEMM -------------------------

enum {
  MODE_BN_RELU_F32 = 0,  // conv1 -> h (fp32 [C,N])
  MODE_STORE_F16T  = 1,  // q -> qT f16 [N,M]
  MODE_BIAS_F16    = 2,  // val + bv -> f16 [M,N]
  MODE_F32         = 3,  // energy -> fp32 [M,N]
  MODE_SUBH_F16T   = 4,  // h - x_r -> dT f16 [N,M]
  MODE_T_FINAL     = 5   // relu(bn(acc+bt)); h+=t; out2=h
};

struct GemmArgs {
  const _Float16* A;   long long a_bstride;   // [M,K] row-major
  const _Float16* Bt;  long long b_bstride;   // [N,K] row-major (B transposed)
  int M, K;                                   // N is always NP
  float*    out_f32; long long o32_bstride;
  _Float16* out_f16; long long o16_bstride;
  const float* hbuf; long long h_bstride;
  float*    hout;
  float*    out2;    long long o2_bstride;
  const float* bias;
  const float* bng; const float* bnb; const float* bnm; const float* bnv;
};

template <int MODE>
__global__ __launch_bounds__(256) void k_gemm(GemmArgs g) {
  // double-buffered staging tiles: K-step 32, row stride 40 halves (80B, 16B aligned)
  __shared__ _Float16 As[2][64 * 40];
  __shared__ _Float16 Bs[2][128 * 40];

  const int bz  = blockIdx.z;
  const int m0  = blockIdx.y * 64;
  const int n0  = blockIdx.x * 128;
  const int tid = threadIdx.x;
  const int lane = tid & 31;
  const int wave = tid >> 5;
  const int wm = (wave >> 2) * 32;   // 0 / 32
  const int wn = (wave & 3)  * 32;   // 0 / 32 / 64 / 96

  const _Float16* A  = g.A  + (long long)bz * g.a_bstride;
  const _Float16* Bt = g.Bt + (long long)bz * g.b_bstride;

  const v8f vzero = {0.f, 0.f, 0.f, 0.f, 0.f, 0.f, 0.f, 0.f};
  v8f acc[2][2];
  acc[0][0] = vzero; acc[0][1] = vzero; acc[1][0] = vzero; acc[1][1] = vzero;

  const int lrow = lane & 15;
  const int kbA  = (lane >> 4) * 8;    // A frag: halves kbA..kbA+7 and kbA+16..kbA+23
  const int kbB  = (lane >> 4) * 8;    // B frag: halves kbB..kbB+7 and kbB+16..kbB+23

  const int arow = tid >> 2, ac4 = tid & 3;   // A tile loader mapping
  const int brow = tid >> 1, bc2 = (tid & 1) * 2;

  // Per-thread DMA destinations (flat low 32 bits of a shared pointer == LDS
  // byte offset on gfx1250); buffer 1 is at a fixed byte delta.
  const unsigned ldsA  = (unsigned)(uintptr_t)(&As[0][arow * 40 + ac4 * 8]);
  const unsigned ldsB0 = (unsigned)(uintptr_t)(&Bs[0][brow * 40 + bc2 * 8]);
  const unsigned ldsB1 = (unsigned)(uintptr_t)(&Bs[0][brow * 40 + (bc2 + 1) * 8]);
  const unsigned dA = (unsigned)(sizeof(As[0]));   // 5120B
  const unsigned dB = (unsigned)(sizeof(Bs[0]));   // 10240B

  const _Float16* srca0 = A + (long long)(m0 + arow) * g.K;
  const _Float16* srcb0 = Bt + (long long)(n0 + brow) * g.K;

  const int steps = g.K >> 5;

  // prologue: stage 0 -> buffer 0
  async_ld_b128(ldsA,  srca0 + ac4 * 8);
  async_ld_b128(ldsB0, srcb0 + bc2 * 8);
  async_ld_b128(ldsB1, srcb0 + (bc2 + 1) * 8);

  for (int s = 0; s < steps; ++s) {
    const int buf = s & 1;
    wait_asynccnt0();     // this wave's stage-s tiles are in LDS
    __syncthreads();      // all waves' tiles visible; all stage s-1 readers done

    if (s + 1 < steps) {  // overlap: DMA stage s+1 into the other buffer
      const int k1 = (s + 1) << 5;
      const unsigned oA = (buf ^ 1) ? dA : 0u;
      const unsigned oB = (buf ^ 1) ? dB : 0u;
      async_ld_b128(ldsA + oA,  srca0 + k1 + ac4 * 8);
      async_ld_b128(ldsB0 + oB, srcb0 + k1 + bc2 * 8);
      async_ld_b128(ldsB1 + oB, srcb0 + k1 + (bc2 + 1) * 8);
      if (s + 2 < steps) {  // L2 prefetch one more stage ahead
        __builtin_prefetch((const void*)(srca0 + k1 + 32 + ac4 * 8), 0, 1);
        __builtin_prefetch((const void*)(srcb0 + k1 + 32 + bc2 * 8), 0, 1);
      }
    }

    v16h af[2], bf[2];
#pragma unroll
    for (int f = 0; f < 2; ++f) {
      const _Float16* pa = &As[buf][(wm + f * 16 + lrow) * 40 + kbA];
      union { v16h v; v8h h[2]; } ua;
      ua.h[0] = *(const v8h*)pa;
      ua.h[1] = *(const v8h*)(pa + 16);
      af[f] = ua.v;
      const _Float16* pb = &Bs[buf][(wn + f * 16 + lrow) * 40 + kbB];
      union { v16h v; v8h h[2]; } ub;
      ub.h[0] = *(const v8h*)pb;
      ub.h[1] = *(const v8h*)(pb + 16);
      bf[f] = ub.v;
    }
#pragma unroll
    for (int i = 0; i < 2; ++i)
#pragma unroll
      for (int j = 0; j < 2; ++j)
        acc[i][j] = __builtin_amdgcn_wmma_f32_16x16x32_f16(
            false, af[i], false, bf[j], (short)0, acc[i][j], false, false);
  }

  // ---- epilogue ----
  float* o32 = nullptr; _Float16* o16 = nullptr;
  const float* hb = nullptr; float* ho = nullptr; float* o2 = nullptr;
  if constexpr (MODE == MODE_BN_RELU_F32 || MODE == MODE_F32)
    o32 = g.out_f32 + (long long)bz * g.o32_bstride;
  if constexpr (MODE == MODE_STORE_F16T || MODE == MODE_BIAS_F16 || MODE == MODE_SUBH_F16T)
    o16 = g.out_f16 + (long long)bz * g.o16_bstride;
  if constexpr (MODE == MODE_SUBH_F16T || MODE == MODE_T_FINAL)
    hb = g.hbuf + (long long)bz * g.h_bstride;
  if constexpr (MODE == MODE_T_FINAL) {
    ho = g.hout + (long long)bz * g.h_bstride;
    o2 = g.out2 + (long long)bz * g.o2_bstride;
  }

  const int ln  = lane & 15;
  const int lm8 = (lane >> 4) * 8;
#pragma unroll
  for (int i = 0; i < 2; ++i) {
#pragma unroll
    for (int j = 0; j < 2; ++j) {
#pragma unroll
      for (int r = 0; r < 8; ++r) {
        int m = m0 + wm + i * 16 + lm8 + r;
        int n = n0 + wn + j * 16 + ln;
        float v = acc[i][j][r];
        if constexpr (MODE == MODE_BN_RELU_F32) {
          float sc = g.bng[m] * rsqrtf(g.bnv[m] + BN_EPS);
          float sh = g.bnb[m] - g.bnm[m] * sc;
          o32[(long long)m * NP + n] = fmaxf(v * sc + sh, 0.0f);
        } else if constexpr (MODE == MODE_STORE_F16T) {
          o16[(long long)n * g.M + m] = (_Float16)v;
        } else if constexpr (MODE == MODE_BIAS_F16) {
          o16[(long long)m * NP + n] = (_Float16)(v + g.bias[m]);
        } else if constexpr (MODE == MODE_F32) {
          o32[(long long)m * NP + n] = v;
        } else if constexpr (MODE == MODE_SUBH_F16T) {
          float d = hb[(long long)m * NP + n] - v;
          o16[(long long)n * g.M + m] = (_Float16)d;
        } else {  // MODE_T_FINAL
          float sc = g.bng[m] * rsqrtf(g.bnv[m] + BN_EPS);
          float sh = g.bnb[m] - g.bnm[m] * sc;
          float t = fmaxf((v + g.bias[m]) * sc + sh, 0.0f);
          float hn = hb[(long long)m * NP + n] + t;
          ho[(long long)m * NP + n] = hn;
          o2[(long long)m * NP + n] = hn;
        }
      }
    }
  }
}

// ------------------------- host launch -------------------------

extern "C" void kernel_launch(void* const* d_in, const int* in_sizes, int n_in,
                              void* d_out, int out_size, void* d_ws, size_t ws_size,
                              hipStream_t stream) {
  const float* x         = (const float*)d_in[0];
  const float* xyz       = (const float*)d_in[1];
  const float* conv1_w   = (const float*)d_in[2];
  const float* convpos_w = (const float*)d_in[3];
  const float* bn1_g = (const float*)d_in[4];
  const float* bn1_b = (const float*)d_in[5];
  const float* bn1_m = (const float*)d_in[6];
  const float* bn1_v = (const float*)d_in[7];
  const float* Wqk = (const float*)d_in[8];
  const float* Wv  = (const float*)d_in[9];
  const float* bv  = (const float*)d_in[10];
  const float* Wt  = (const float*)d_in[11];
  const float* bt  = (const float*)d_in[12];
  const float* bng = (const float*)d_in[13];
  const float* bnb = (const float*)d_in[14];
  const float* bnm = (const float*)d_in[15];
  const float* bnv = (const float*)d_in[16];
  float* out = (float*)d_out;

  char* p = (char*)d_ws;
  auto carve = [&](size_t bytes) -> char* {
    char* r = p;
    p += (bytes + 255) & ~(size_t)255;
    return r;
  };
  _Float16* w16_c1 = (_Float16*)carve((size_t)CC * CC * 2);
  _Float16* w16_qk = (_Float16*)carve((size_t)4 * DD * CC * 2);
  _Float16* w16_v  = (_Float16*)carve((size_t)4 * CC * CC * 2);
  _Float16* w16_t  = (_Float16*)carve((size_t)4 * CC * CC * 2);
  float*    pos    = (float*)carve((size_t)BB * CC * NP * 4);
  float*    hbuf   = (float*)carve((size_t)BB * CC * NP * 4);
  _Float16* xT     = (_Float16*)carve((size_t)BB * NP * CC * 2);
  _Float16* h2T    = (_Float16*)carve((size_t)BB * NP * CC * 2);
  _Float16* qT     = (_Float16*)carve((size_t)BB * NP * DD * 2);
  _Float16* val16  = (_Float16*)carve((size_t)BB * CC * NP * 2);
  _Float16* dT     = (_Float16*)carve((size_t)BB * NP * CC * 2);
  float*    energy = (float*)carve((size_t)BB * NP * NP * 4);
  float*    rmax   = (float*)carve((size_t)BB * NP * 4);
  float*    rsum   = (float*)carve((size_t)BB * NP * 4);
  _Float16* attT   = (_Float16*)carve((size_t)BB * NP * NP * 2);

  k_cvt_f16<<<(CC * CC + 255) / 256, 256, 0, stream>>>(conv1_w, w16_c1, CC * CC);
  k_cvt_f16<<<(4 * DD * CC + 255) / 256, 256, 0, stream>>>(Wqk, w16_qk, 4 * DD * CC);
  k_cvt_f16<<<(4 * CC * CC + 255) / 256, 256, 0, stream>>>(Wv, w16_v, 4 * CC * CC);
  k_cvt_f16<<<(4 * CC * CC + 255) / 256, 256, 0, stream>>>(Wt, w16_t, 4 * CC * CC);

  long long tot = (long long)BB * CC * NP;
  int blk = (int)((tot + 255) / 256);
  k_pos<<<blk, 256, 0, stream>>>(convpos_w, xyz, pos);
  k_transpose_cvt<<<blk, 256, 0, stream>>>(x, xT);

  // h = relu(bn1(conv1_w @ x))
  {
    GemmArgs g{};
    g.A = w16_c1; g.a_bstride = 0;
    g.Bt = xT;    g.b_bstride = (long long)NP * CC;
    g.M = CC; g.K = CC;
    g.out_f32 = hbuf; g.o32_bstride = (long long)CC * NP;
    g.bng = bn1_g; g.bnb = bn1_b; g.bnm = bn1_m; g.bnv = bn1_v;
    k_gemm<MODE_BN_RELU_F32><<<dim3(NP / 128, CC / 64, BB), 256, 0, stream>>>(g);
  }

  for (int i = 0; i < 4; ++i) {
    k_h2T<<<blk, 256, 0, stream>>>(hbuf, pos, h2T);
    {  // q = Wqk @ h2  -> qT [N,64]
      GemmArgs g{};
      g.A = w16_qk + (size_t)i * DD * CC; g.a_bstride = 0;
      g.Bt = h2T; g.b_bstride = (long long)NP * CC;
      g.M = DD; g.K = CC;
      g.out_f16 = qT; g.o16_bstride = (long long)NP * DD;
      k_gemm<MODE_STORE_F16T><<<dim3(NP / 128, 1, BB), 256, 0, stream>>>(g);
    }
    {  // val = Wv @ h2 + bv -> [C,N]
      GemmArgs g{};
      g.A = w16_v + (size_t)i * CC * CC; g.a_bstride = 0;
      g.Bt = h2T; g.b_bstride = (long long)NP * CC;
      g.M = CC; g.K = CC;
      g.out_f16 = val16; g.o16_bstride = (long long)CC * NP;
      g.bias = bv + i * CC;
      k_gemm<MODE_BIAS_F16><<<dim3(NP / 128, CC / 64, BB), 256, 0, stream>>>(g);
    }
    {  // energy[n,m] = sum_d qT[n,d]*qT[m,d]
      GemmArgs g{};
      g.A = qT;  g.a_bstride = (long long)NP * DD;
      g.Bt = qT; g.b_bstride = (long long)NP * DD;
      g.M = NP; g.K = DD;
      g.out_f32 = energy; g.o32_bstride = (long long)NP * NP;
      k_gemm<MODE_F32><<<dim3(NP / 128, NP / 64, BB), 256, 0, stream>>>(g);
    }
    k_rowstat<<<dim3(NP, BB), 256, 0, stream>>>(energy, rmax, rsum);
    k_colnorm<<<dim3(NP, BB), 256, 0, stream>>>(energy, rmax, rsum, attT);
    {  // x_r = val @ att ; dT[n,c] = f16(h - x_r)
      GemmArgs g{};
      g.A = val16; g.a_bstride = (long long)CC * NP;
      g.Bt = attT; g.b_bstride = (long long)NP * NP;
      g.M = CC; g.K = NP;
      g.hbuf = hbuf; g.h_bstride = (long long)CC * NP;
      g.out_f16 = dT; g.o16_bstride = (long long)NP * CC;
      k_gemm<MODE_SUBH_F16T><<<dim3(NP / 128, CC / 64, BB), 256, 0, stream>>>(g);
    }
    {  // t = relu(bn(Wt @ d + bt)); h += t; out chunk = h
      GemmArgs g{};
      g.A = w16_t + (size_t)i * CC * CC; g.a_bstride = 0;
      g.Bt = dT; g.b_bstride = (long long)NP * CC;
      g.M = CC; g.K = CC;
      g.hbuf = hbuf; g.hout = hbuf; g.h_bstride = (long long)CC * NP;
      g.out2 = out + (size_t)i * CC * NP; g.o2_bstride = (long long)4 * CC * NP;
      g.bias = bt + i * CC;
      g.bng = bng + i * CC; g.bnb = bnb + i * CC;
      g.bnm = bnm + i * CC; g.bnv = bnv + i * CC;
      k_gemm<MODE_T_FINAL><<<dim3(NP / 128, CC / 64, BB), 256, 0, stream>>>(g);
    }
  }
}
